// SpectralTransposed2d_32736240730399
// MI455X (gfx1250) — matrix operand order repaired and checked by reading, not compiled, not run on hardware
//
#include <hip/hip_runtime.h>
#include <hip/hip_bf16.h>

typedef __attribute__((ext_vector_type(16))) __bf16 v16bf;
typedef __attribute__((ext_vector_type(8)))  float  v8f;

#define PI_F 3.14159265358979323846f

// ---------------------------------------------------------------------------
// Single-row in-LDS 256-point radix-2 complex FFT, 128 threads.
// dir = -1 forward (numpy convention), +1 inverse (unnormalized).
// ---------------------------------------------------------------------------
__device__ inline void fft256_lds(float* re, float* im, float dir) {
  const int tid = threadIdx.x; // 0..127
  for (int k = tid; k < 256; k += 128) {
    int r = __brev((unsigned)k) >> 24;
    if (r > k) {
      float tr = re[k]; re[k] = re[r]; re[r] = tr;
      float ti = im[k]; im[k] = im[r]; im[r] = ti;
    }
  }
  __syncthreads();
  #pragma unroll
  for (int s = 1; s <= 8; ++s) {
    const int m = 1 << s, half = m >> 1;
    const int grp = tid >> (s - 1);
    const int pos = tid & (half - 1);
    const int i1 = grp * m + pos;
    const int i2 = i1 + half;
    const float ang = dir * 2.0f * PI_F * (float)pos / (float)m;
    float sn, c;
    __sincosf(ang, &sn, &c);
    const float xr = re[i2], xi = im[i2];
    const float tr = xr * c - xi * sn;
    const float ti = xr * sn + xi * c;
    const float ur = re[i1], ui = im[i1];
    re[i1] = ur + tr; im[i1] = ui + ti;
    re[i2] = ur - tr; im[i2] = ui - ti;
    __syncthreads();
  }
}

// ---------------------------------------------------------------------------
// 16 simultaneous 256-point FFTs down the columns of a [256][17]-padded LDS
// tile (pad => 16 consecutive columns hit 16 distinct banks). 128 threads,
// 2048 butterflies/stage (16 per thread).
// ---------------------------------------------------------------------------
__device__ inline void fft256_cols16(float (*re)[17], float (*im)[17], float dir) {
  const int tid = threadIdx.x;
  for (int idx = tid; idx < 256 * 16; idx += 128) {
    const int c = idx & 15, k = idx >> 4;
    const int r = __brev((unsigned)k) >> 24;
    if (r > k) {
      float tr = re[k][c]; re[k][c] = re[r][c]; re[r][c] = tr;
      float ti = im[k][c]; im[k][c] = im[r][c]; im[r][c] = ti;
    }
  }
  __syncthreads();
  #pragma unroll
  for (int s = 1; s <= 8; ++s) {
    const int m = 1 << s, half = m >> 1;
    for (int idx = tid; idx < 128 * 16; idx += 128) {
      const int c = idx & 15, j = idx >> 4;
      const int grp = j >> (s - 1);
      const int pos = j & (half - 1);
      const int i1 = grp * m + pos;
      const int i2 = i1 + half;
      const float ang = dir * 2.0f * PI_F * (float)pos / (float)m;
      float sn, co;
      __sincosf(ang, &sn, &co);
      const float xr = re[i2][c], xi = im[i2][c];
      const float tr = xr * co - xi * sn;
      const float ti = xr * sn + xi * co;
      const float ur = re[i1][c], ui = im[i1][c];
      re[i1][c] = ur + tr; im[i1][c] = ui + ti;
      re[i2][c] = ur - tr; im[i2][c] = ui - ti;
    }
    __syncthreads();
  }
}

// ---------------------------------------------------------------------------
// Kernel 1: row rfft over W=256 for x[b][o][h][:], keep bins 0..128.
// X_fft layout: float2 at ((b*64+o)*256+h)*129 + wf  (both sides coalesced)
// ---------------------------------------------------------------------------
__global__ __launch_bounds__(128) void k_row_rfft(const float* __restrict__ x,
                                                  float2* __restrict__ Xf) {
  __shared__ float re[256], im[256];
  const int h = blockIdx.x, o = blockIdx.y, b = blockIdx.z;
  const size_t row = ((size_t)b * 64 + o) * 256 + h;
  const float* src = x + row * 256;
  for (int k = threadIdx.x; k < 256; k += 128) { re[k] = src[k]; im[k] = 0.0f; }
  __syncthreads();
  fft256_lds(re, im, -1.0f);
  float2* dst = Xf + row * 129;
  const int t = threadIdx.x;
  dst[t] = make_float2(re[t], im[t]);
  if (t == 0) dst[128] = make_float2(re[128], im[128]);
}

// ---------------------------------------------------------------------------
// Kernel 2: forward complex FFT over H; block owns 16 adjacent wf columns of
// one (b,o) plane so each row access is 16 consecutive float2 = 128B.
// ---------------------------------------------------------------------------
__global__ __launch_bounds__(128) void k_col_fft(float2* __restrict__ Xf) {
  __shared__ float re[256][17], im[256][17];
  const int wf0 = blockIdx.x * 16, o = blockIdx.y, b = blockIdx.z;
  const size_t base = ((size_t)b * 64 + o) * 256 * 129;
  for (int idx = threadIdx.x; idx < 256 * 16; idx += 128) {
    const int c = idx & 15, h = idx >> 4;
    const int wf = wf0 + c;
    if (wf < 129) {
      float2 v = Xf[base + (size_t)h * 129 + wf];
      re[h][c] = v.x; im[h][c] = v.y;
    } else { re[h][c] = 0.0f; im[h][c] = 0.0f; }
  }
  __syncthreads();
  fft256_cols16(re, im, -1.0f);
  for (int idx = threadIdx.x; idx < 256 * 16; idx += 128) {
    const int c = idx & 15, h = idx >> 4;
    const int wf = wf0 + c;
    if (wf < 129) Xf[base + (size_t)h * 129 + wf] = make_float2(re[h][c], im[h][c]);
  }
}

// ---------------------------------------------------------------------------
// Kernel 3: per-frequency-bin GEMM via WMMA (unchanged from round 1).
//   A (16x64 bf16): rows 0-7 = Re X[b][o], rows 8-15 = Im X[b][o]  (K = o)
//   B (64x64 bf16): W[o][i] for this bin
//   C (16x64 f32) -> Z[h][wf][m][i]
// ---------------------------------------------------------------------------
__global__ __launch_bounds__(128) void k_bin_gemm_wmma(const float2* __restrict__ Xf,
                                                       const float* __restrict__ Wf,
                                                       float* __restrict__ Z) {
  __shared__ __bf16 Wt[64 * 16 * 64]; // [i][t][o]  128 KB
  __shared__ __bf16 Xt[16 * 16 * 64]; // [t][m][o]   32 KB
  const int wf0 = blockIdx.x * 16;
  const int h   = blockIdx.y;
  const int tid = threadIdx.x;

  for (int p = tid; p < 64 * 64; p += 128) {
    const int o = p >> 6, i = p & 63;
    const size_t gbase = (((size_t)o * 64 + i) * 256 + h) * 129;
    #pragma unroll
    for (int t = 0; t < 16; ++t) {
      int wf = wf0 + t; if (wf > 128) wf = 128;
      Wt[((i * 16 + t) << 6) + o] = (__bf16)Wf[gbase + (size_t)wf];
    }
  }
  for (int p = tid; p < 8 * 64 * 16; p += 128) {
    const int b = p >> 10, o = (p >> 4) & 63, t = p & 15;
    int wf = wf0 + t; if (wf > 128) wf = 128;
    float2 v = Xf[(((size_t)b * 64 + o) * 256 + h) * 129 + (size_t)wf];
    Xt[((t * 16 + b) << 6) + o]     = (__bf16)v.x;
    Xt[((t * 16 + 8 + b) << 6) + o] = (__bf16)v.y;
  }
  __syncthreads();

  const int wave = tid >> 5, lane = tid & 31;
  const int hi = lane >> 4, lrow = lane & 15;

  for (int q = 0; q < 4; ++q) {
    const int t = wave * 4 + q;
    v16bf a0, a1;
    const int xb = (t * 16 + lrow) << 6;
    #pragma unroll
    for (int j = 0; j < 8; ++j) {
      const int ob = ((j >> 2) << 4) + (hi << 3) + ((j & 3) << 1);
      a0[2 * j]     = Xt[xb + ob];
      a0[2 * j + 1] = Xt[xb + ob + 1];
      a1[2 * j]     = Xt[xb + 32 + ob];
      a1[2 * j + 1] = Xt[xb + 32 + ob + 1];
    }
    const int wf = wf0 + t;
    #pragma unroll
    for (int nt = 0; nt < 4; ++nt) {
      v8f acc = {};
      v16bf b0, b1;
      const int wb = (((nt * 16 + lrow) * 16 + t) << 6) + (hi << 3);
      #pragma unroll
      for (int e = 0; e < 16; ++e) {
        b0[e] = Wt[wb + e + ((e >> 3) << 3)];
        b1[e] = Wt[wb + 32 + e + ((e >> 3) << 3)];
      }
      acc = __builtin_amdgcn_wmma_f32_16x16x32_bf16(false, a0, false, b0,
                                                    (short)0, acc, false, false);
      acc = __builtin_amdgcn_wmma_f32_16x16x32_bf16(false, a1, false, b1,
                                                    (short)0, acc, false, false);
      if (wf < 129) {
        float* zb = Z + (((size_t)h * 129 + wf) * 16) * 64 + nt * 16 + lrow;
        #pragma unroll
        for (int r = 0; r < 8; ++r) {
          const int m = (hi << 3) + r;
          zb[(size_t)m * 64] = acc[r];
        }
      }
    }
  }
}

// ---------------------------------------------------------------------------
// Kernel 4: inverse complex FFT over H on Z[h][wf][m][i], in place.
// Block owns (wf, one b, 16 consecutive i) = 16 complex columns; every
// (h,wf,m) row touched as a contiguous 64B segment.
// ---------------------------------------------------------------------------
__global__ __launch_bounds__(128) void k_col_ifft_z(float* __restrict__ Z) {
  __shared__ float re[256][17], im[256][17];
  const int wf = blockIdx.x, i0 = blockIdx.y * 16, b = blockIdx.z;
  for (int idx = threadIdx.x; idx < 256 * 16; idx += 128) {
    const int c = idx & 15, h = idx >> 4;
    const size_t base = ((size_t)h * 129 + wf) * 16;
    re[h][c] = Z[(base + b) * 64 + i0 + c];
    im[h][c] = Z[(base + 8 + b) * 64 + i0 + c];
  }
  __syncthreads();
  fft256_cols16(re, im, +1.0f);
  for (int idx = threadIdx.x; idx < 256 * 16; idx += 128) {
    const int c = idx & 15, h = idx >> 4;
    const size_t base = ((size_t)h * 129 + wf) * 16;
    Z[(base + b) * 64 + i0 + c]     = re[h][c];
    Z[(base + 8 + b) * 64 + i0 + c] = im[h][c];
  }
}

// ---------------------------------------------------------------------------
// Kernel 5: Hermitian-extend 129 -> 256 bins, inverse row FFT, real part,
// scale 1/(256*256), add bias. Block owns (h, one b, 16 consecutive i):
// bin loads are 64B rows, output stores are fully coalesced along w.
// ---------------------------------------------------------------------------
__global__ __launch_bounds__(128) void k_row_irfft(const float* __restrict__ Z,
                                                   const float* __restrict__ bias,
                                                   float* __restrict__ y) {
  __shared__ float re[256][17], im[256][17];
  const int h = blockIdx.x, i0 = blockIdx.y * 16, b = blockIdx.z;
  const int tid = threadIdx.x;
  for (int idx = tid; idx < 129 * 16; idx += 128) {
    const int c = idx & 15, wf = idx >> 4;
    const size_t base = ((size_t)h * 129 + wf) * 16;
    re[wf][c] = Z[(base + b) * 64 + i0 + c];
    im[wf][c] = Z[(base + 8 + b) * 64 + i0 + c];
  }
  __syncthreads();
  for (int idx = tid; idx < 127 * 16; idx += 128) { // mirror 1..127 -> 255..129
    const int c = idx & 15, wf = (idx >> 4) + 1;
    re[256 - wf][c] = re[wf][c];
    im[256 - wf][c] = -im[wf][c];
  }
  __syncthreads();
  fft256_cols16(re, im, +1.0f);
  for (int c = 0; c < 16; ++c) {
    const int i = i0 + c;
    const float bi = bias[i];
    float* dst = y + (((size_t)b * 64 + i) * 256 + h) * 256;
    for (int w = tid; w < 256; w += 128)
      dst[w] = re[w][c] * (1.0f / 65536.0f) + bi;
  }
}

// ---------------------------------------------------------------------------
extern "C" void kernel_launch(void* const* d_in, const int* in_sizes, int n_in,
                              void* d_out, int out_size, void* d_ws, size_t ws_size,
                              hipStream_t stream) {
  (void)in_sizes; (void)n_in; (void)out_size; (void)ws_size;
  const float* x    = (const float*)d_in[0]; // (8,64,256,256)
  const float* wfft = (const float*)d_in[1]; // (64,64,256,129)
  const float* bias = (const float*)d_in[2]; // (64,)
  float* out = (float*)d_out;                // (8,64,256,256)

  float2* Xf = (float2*)d_ws;
  float*  Zp = (float*)((char*)d_ws + (size_t)8 * 64 * 256 * 129 * sizeof(float2));

  const dim3 blk(128);
  k_row_rfft     <<<dim3(256, 64, 8), blk, 0, stream>>>(x, Xf);
  k_col_fft      <<<dim3(9, 64, 8),   blk, 0, stream>>>(Xf);
  k_bin_gemm_wmma<<<dim3(9, 256),     blk, 0, stream>>>(Xf, wfft, Zp);
  k_col_ifft_z   <<<dim3(129, 4, 8),  blk, 0, stream>>>(Zp);
  k_row_irfft    <<<dim3(256, 4, 8),  blk, 0, stream>>>(Zp, bias, out);
}
